// FNOBlockNd_35218731827945
// MI455X (gfx1250) — matrix operand order repaired and checked by reading, not compiled, not run on hardware
//
#include <hip/hip_runtime.h>
#include <hip/hip_bf16.h>
#include <math.h>

// ---------------------------------------------------------------------------
// FNO block for MI455X (gfx1250, wave32, WMMA + TDM).
// Truncated-mode FFTs expressed as dense DFT GEMMs -> all heavy math on
// v_wmma_f32_16x16x32_f16; U operand staged to LDS via tensor_load_to_lds.
// ---------------------------------------------------------------------------

typedef __attribute__((ext_vector_type(16))) _Float16 v16h;
typedef __attribute__((ext_vector_type(8)))  float    v8f;

#if __has_builtin(__builtin_amdgcn_tensor_load_to_lds)
#define HAVE_TDM 1
typedef __attribute__((ext_vector_type(4))) unsigned int v4u;
typedef __attribute__((ext_vector_type(8))) int          v8i;
typedef __attribute__((ext_vector_type(4))) int          v4i;
#else
#define HAVE_TDM 0
#endif

#define Bsz   16
#define CIN   64
#define COUT  64
#define Hsz   256
#define Wsz   256
#define M0    20
#define M1    20
#define HW    (Hsz * Wsz)          // 65536
#define NKL   (M0 * M1)            // 400
#define NBI   (Bsz * CIN)          // 1024
#define NBO   (Bsz * COUT)         // 1024

// Workspace layout (float elements).  T (48x256 f32 per bi) and U (64x256 f16
// per bo) alias the same region: T is fully consumed by k2 before k4 writes U.
#define OFF_FHS   ((size_t)0)                      // 48*256          = 12288
#define OFF_FWRE  (OFF_FHS  + 48 * 256)            // 20*256          =  5120
#define OFF_FWIM  (OFF_FWRE + 20 * 256)
#define OFF_GWRE  (OFF_FWIM + 20 * 256)
#define OFF_GWIM  (OFF_GWRE + 20 * 256)
#define OFF_GHS   (OFF_GWIM + 20 * 256)            // 256*64          = 16384
#define OFF_XFR   (OFF_GHS  + 256 * 64)            // 400*1024        = 409600
#define OFF_XFI   (OFF_XFR  + (size_t)NKL * NBI)
#define OFF_YFR   (OFF_XFI  + (size_t)NKL * NBI)
#define OFF_YFI   (OFF_YFR  + (size_t)NKL * NBO)
#define OFF_TU    (OFF_YFI  + (size_t)NKL * NBO)   // T: 1024*48*256 f32

// ---------------------------------------------------------------------------
// WMMA helpers (wave32; A and B both use contiguous k = 16*(lane/16) + e)
// ---------------------------------------------------------------------------
__device__ __forceinline__ v8f wmma_f16(v16h a, v16h b, v8f c) {
  return __builtin_amdgcn_wmma_f32_16x16x32_f16(
      /*neg_a=*/false, a, /*neg_b=*/false, b,
      /*c_mod=*/(short)0, c, /*reuse_a=*/false, /*reuse_b=*/false);
}

// A fragment (16x32 f16, M x K): lanes 0-15 hold rows M=lane, K=0..15;
// lanes 16-31 same rows, K=16..31.
__device__ __forceinline__ v16h load_a16(const _Float16* A, int lda,
                                         int mBase, int kBase, int lane) {
  v16h a;
  const int m    = mBase + (lane & 15);
  const int kadj = (lane >> 4) << 4;
#pragma unroll
  for (int e = 0; e < 16; ++e) {
    const int k = kBase + kadj + e;
    a[e] = A[m * lda + k];
  }
  return a;
}

// B fragment (32x16 f16, K x N): lanes 0-15 hold col N=lane, K=0..15;
// lanes 16-31 hold same cols, K=16..31.
__device__ __forceinline__ v16h load_b16(const _Float16* Bm, int ldb,
                                         int kBase, int nBase, int lane) {
  v16h b;
  const int n  = nBase + (lane & 15);
  const int k0 = kBase + ((lane >> 4) << 4);
#pragma unroll
  for (int e = 0; e < 16; ++e) b[e] = Bm[(k0 + e) * ldb + n];
  return b;
}

__device__ __forceinline__ float gelu_f(float v) {
  const float c = 0.7978845608028654f * (v + 0.044715f * v * v * v);
  return 0.5f * v * (1.0f + tanhf(c));
}

// ---------------------------------------------------------------------------
// k0: DFT twiddle factor tables
// ---------------------------------------------------------------------------
__global__ __launch_bounds__(256) void k0_twiddles(float* __restrict__ ws) {
  const float PI2 = 6.283185307179586f;
  const float s   = 0.0625f;  // 1/sqrt(256) per transform stage
  const int stride = gridDim.x * blockDim.x;
  const int tid = blockIdx.x * blockDim.x + threadIdx.x;

  // Fhs[48][256]: rows 2k = cos, 2k+1 = -sin (forward H-axis), pad to 48.
  for (int t = tid; t < 48 * 256; t += stride) {
    const int m = t >> 8, h = t & 255;
    float v = 0.f;
    if (m < 2 * M0) {
      const int k = m >> 1;
      const float ang = PI2 * (float)(k * h) * (1.0f / 256.0f);
      v = ((m & 1) ? -__sinf(ang) : __cosf(ang)) * s;
    }
    ws[OFF_FHS + t] = v;
  }
  // Fw / Gw (W axis, forward / inverse with hermitian doubling).
  for (int t = tid; t < M1 * 256; t += stride) {
    const int l = t >> 8, w = t & 255;
    const float ang = PI2 * (float)(l * w) * (1.0f / 256.0f);
    const float cs = __cosf(ang), sn = __sinf(ang);
    ws[OFF_FWRE + t] = cs * s;
    ws[OFF_FWIM + t] = -sn * s;
    const float dbl = (l == 0) ? 1.0f : 2.0f;
    ws[OFF_GWRE + t] = dbl * cs * s;
    ws[OFF_GWIM + t] = dbl * sn * s;
  }
  // Ghs[256][64]: cols 2k = cos, 2k+1 = -sin (inverse H-axis), pad to 64.
  for (int t = tid; t < 256 * 64; t += stride) {
    const int h = t >> 6, m = t & 63;
    float v = 0.f;
    if (m < 2 * M0) {
      const int k = m >> 1;
      const float ang = PI2 * (float)(k * h) * (1.0f / 256.0f);
      v = ((m & 1) ? -__sinf(ang) : __cosf(ang)) * s;
    }
    ws[OFF_GHS + t] = v;
  }
}

// ---------------------------------------------------------------------------
// k1: forward row DFT  T[bi](48x256) = Fhs(48x256) @ x[bi](256x256)   [WMMA]
// ---------------------------------------------------------------------------
__global__ __launch_bounds__(256) void k1_rowdft(const float* __restrict__ x,
                                                 float* __restrict__ ws) {
  __shared__ _Float16 Ash[48 * 256];  // 24 KB
  for (int t = threadIdx.x; t < 48 * 256; t += 256)
    Ash[t] = (_Float16)ws[OFF_FHS + t];
  __syncthreads();

  const int bi   = blockIdx.x >> 1;
  const int half = blockIdx.x & 1;
  const int wave = threadIdx.x >> 5, lane = threadIdx.x & 31;
  const int n0 = half * 128 + wave * 16;
  const float* __restrict__ Xb = x + (size_t)bi * HW;

  v8f acc0 = {}, acc1 = {}, acc2 = {};
#pragma unroll
  for (int kb = 0; kb < 8; ++kb) {
    const int n = n0 + (lane & 15);
    const int k0 = kb * 32 + ((lane >> 4) << 4);
    if (kb < 7)
      __builtin_prefetch(&Xb[(size_t)((kb + 1) * 32) * 256 + n], 0, 1);
    v16h bf;
#pragma unroll
    for (int e = 0; e < 16; ++e)
      bf[e] = (_Float16)Xb[(size_t)(k0 + e) * 256 + n];
    v16h a0 = load_a16(Ash, 256, 0, kb * 32, lane);
    v16h a1 = load_a16(Ash, 256, 16, kb * 32, lane);
    v16h a2 = load_a16(Ash, 256, 32, kb * 32, lane);
    acc0 = wmma_f16(a0, bf, acc0);
    acc1 = wmma_f16(a1, bf, acc1);
    acc2 = wmma_f16(a2, bf, acc2);
  }

  float* __restrict__ Tb = ws + OFF_TU + (size_t)bi * 48 * 256;
#pragma unroll
  for (int r = 0; r < 8; ++r) {
    const int m = r + ((lane >> 4) << 3);
    const int n = n0 + (lane & 15);
    Tb[(m + 0) * 256 + n]  = acc0[r];
    Tb[(m + 16) * 256 + n] = acc1[r];
    Tb[(m + 32) * 256 + n] = acc2[r];
  }
}

// ---------------------------------------------------------------------------
// k2: forward col DFT + truncate:  Xf[kl][bi] = sum_w T[bi](k,:) * Fw(l,:)
// ---------------------------------------------------------------------------
__global__ __launch_bounds__(256) void k2_coldft(float* __restrict__ ws) {
  __shared__ float Tsh[40 * 256];  // 40 KB
  const int bi = blockIdx.x;
  const float* __restrict__ Tb = ws + OFF_TU + (size_t)bi * 48 * 256;
  for (int t = threadIdx.x; t < 40 * 256; t += 256) Tsh[t] = Tb[t];
  __syncthreads();

  for (int kl = threadIdx.x; kl < NKL; kl += 256) {
    const int k = kl / M1, l = kl % M1;
    const float* __restrict__ fr = ws + OFF_FWRE + l * 256;
    const float* __restrict__ fi = ws + OFF_FWIM + l * 256;
    const float* __restrict__ tr = Tsh + (2 * k) * 256;
    const float* __restrict__ ti = Tsh + (2 * k + 1) * 256;
    float xr = 0.f, xi = 0.f;
#pragma unroll 4
    for (int w = 0; w < 256; ++w) {
      const float a = tr[w], b = ti[w], c = fr[w], d = fi[w];
      xr += a * c - b * d;
      xi += a * d + b * c;
    }
    ws[OFF_XFR + (size_t)kl * NBI + bi] = xr;
    ws[OFF_XFI + (size_t)kl * NBI + bi] = xi;
  }
}

// ---------------------------------------------------------------------------
// k3: per-mode complex channel mix (b,i)->(b,o)                        [WMMA]
// ---------------------------------------------------------------------------
__global__ __launch_bounds__(128) void k3_mix(float* __restrict__ ws,
                                              const float* __restrict__ w_real,
                                              const float* __restrict__ w_imag) {
  __shared__ _Float16 XRs[16 * 64], XIs[16 * 64];
  __shared__ _Float16 WRs[64 * 64], WIs[64 * 64];
  const int kl = blockIdx.x, tid = threadIdx.x;

  for (int t = tid; t < 16 * 64; t += 128) {
    XRs[t] = (_Float16)ws[OFF_XFR + (size_t)kl * NBI + t];
    XIs[t] = (_Float16)ws[OFF_XFI + (size_t)kl * NBI + t];
  }
  for (int t = tid; t < 64 * 64; t += 128) {
    const int i = t >> 6, o = t & 63;
    WRs[t] = (_Float16)w_real[(size_t)(i * COUT + o) * NKL + kl];
    WIs[t] = (_Float16)w_imag[(size_t)(i * COUT + o) * NKL + kl];
  }
  __syncthreads();

  const int wave = tid >> 5, lane = tid & 31;
  const int n0 = wave << 4;
  v8f p1 = {}, p2 = {}, p3 = {}, p4 = {};
#pragma unroll
  for (int kb = 0; kb < 2; ++kb) {
    v16h aR = load_a16(XRs, 64, 0, kb * 32, lane);
    v16h aI = load_a16(XIs, 64, 0, kb * 32, lane);
    v16h bR = load_b16(WRs, 64, kb * 32, n0, lane);
    v16h bI = load_b16(WIs, 64, kb * 32, n0, lane);
    p1 = wmma_f16(aR, bR, p1);  // Re*Re
    p2 = wmma_f16(aI, bI, p2);  // Im*Im
    p3 = wmma_f16(aR, bI, p3);  // Re*Im
    p4 = wmma_f16(aI, bR, p4);  // Im*Re
  }
#pragma unroll
  for (int r = 0; r < 8; ++r) {
    const int b = r + ((lane >> 4) << 3);
    const int o = n0 + (lane & 15);
    ws[OFF_YFR + (size_t)kl * NBO + b * COUT + o] = p1[r] - p2[r];
    ws[OFF_YFI + (size_t)kl * NBO + b * COUT + o] = p3[r] + p4[r];
  }
}

// ---------------------------------------------------------------------------
// k4: inverse col DFT:  U[bo](64x256 f16; rows 2k/2k+1 = Re/Im, pad = 0)
// U stored f16 so k5b can TDM it straight into LDS as a WMMA operand.
// ---------------------------------------------------------------------------
__global__ __launch_bounds__(256) void k4_invcol(float* __restrict__ ws) {
  __shared__ float YR[NKL], YI[NKL];
  const int bo = blockIdx.x, tid = threadIdx.x;
  for (int t = tid; t < NKL; t += 256) {
    YR[t] = ws[OFF_YFR + (size_t)t * NBO + bo];
    YI[t] = ws[OFF_YFI + (size_t)t * NBO + bo];
  }
  __syncthreads();

  const int w = tid;
  _Float16* __restrict__ Ub =
      (_Float16*)(ws + OFF_TU) + (size_t)bo * 64 * 256;
#pragma unroll 2
  for (int k = 0; k < M0; ++k) {
    float ur = 0.f, ui = 0.f;
#pragma unroll
    for (int l = 0; l < M1; ++l) {
      const float gr = ws[OFF_GWRE + l * 256 + w];
      const float gi = ws[OFF_GWIM + l * 256 + w];
      const float yr = YR[k * M1 + l], yi = YI[k * M1 + l];
      ur += yr * gr - yi * gi;
      ui += yr * gi + yi * gr;
    }
    Ub[(2 * k) * 256 + w]     = (_Float16)ur;
    Ub[(2 * k + 1) * 256 + w] = (_Float16)ui;
  }
#pragma unroll
  for (int m = 2 * M0; m < 64; ++m) Ub[m * 256 + w] = (_Float16)0.f;
}

// ---------------------------------------------------------------------------
// k5a: bypass 1x1 conv:  out[b] = conv_w(64x64) @ x[b](64x65536) + bias [WMMA]
// ---------------------------------------------------------------------------
__global__ __launch_bounds__(256) void k5a_bypass(const float* __restrict__ x,
                                                  const float* __restrict__ conv_w,
                                                  const float* __restrict__ conv_b,
                                                  float* __restrict__ out) {
  __shared__ _Float16 Ash[64 * 64];  // conv_w[o][i], 8 KB
  const int b = blockIdx.x >> 9, tile = blockIdx.x & 511;
  const int tid = threadIdx.x, wave = tid >> 5, lane = tid & 31;
  for (int t = tid; t < 64 * 64; t += 256) Ash[t] = (_Float16)conv_w[t];
  __syncthreads();

  const size_t n0 = (size_t)tile * 128 + wave * 16;
  const float* __restrict__ Xb = x + (size_t)b * CIN * HW;
  v8f acc[4] = {};
#pragma unroll
  for (int kb = 0; kb < 2; ++kb) {
    const int i0 = kb * 32 + ((lane >> 4) << 4);
    const size_t n = n0 + (lane & 15);
    __builtin_prefetch(&Xb[(size_t)i0 * HW + n + 128], 0, 1);
    v16h bf;
#pragma unroll
    for (int e = 0; e < 16; ++e)
      bf[e] = (_Float16)Xb[(size_t)(i0 + e) * HW + n];
#pragma unroll
    for (int mt = 0; mt < 4; ++mt) {
      v16h a = load_a16(Ash, 64, mt * 16, kb * 32, lane);
      acc[mt] = wmma_f16(a, bf, acc[mt]);
    }
  }

  float* __restrict__ Ob = out + (size_t)b * COUT * HW;
#pragma unroll
  for (int mt = 0; mt < 4; ++mt) {
#pragma unroll
    for (int r = 0; r < 8; ++r) {
      const int o = mt * 16 + r + ((lane >> 4) << 3);
      const size_t hw = n0 + (lane & 15);
      Ob[(size_t)o * HW + hw] = acc[mt][r] + conv_b[o];
    }
  }
}

// ---------------------------------------------------------------------------
// k5b: inverse row DFT + residual + GELU                         [WMMA + TDM]
// S(16x256 tile) = Ghs[h-tile](16x64) @ U[bo](64x256 f16); out = gelu(S+byp).
// U tile (32 KB contiguous f16) staged to LDS with TENSOR_LOAD_TO_LDS.
// ---------------------------------------------------------------------------
__global__ __launch_bounds__(256) void k5b_final(const float* __restrict__ ws,
                                                 float* __restrict__ out) {
  __shared__ _Float16 Ash[16 * 64];   // 2 KB
  __shared__ _Float16 Bsh[64 * 256];  // 32 KB
  const int bo = blockIdx.x >> 4, mt = blockIdx.x & 15;
  const int tid = threadIdx.x, wave = tid >> 5, lane = tid & 31;
  const int h0 = mt * 16;

  const _Float16* __restrict__ Ub =
      (const _Float16*)(ws + OFF_TU) + (size_t)bo * 64 * 256;

#if HAVE_TDM
  if (tid < 32) {
    // Tensor DMA descriptor (ISA 8.3/8.4): 2-D tile, 64 rows x 256 cols of
    // 2-byte elements, contiguous (dim0_stride = 256), into LDS at Bsh.
    const unsigned long long ga = (unsigned long long)(uintptr_t)Ub;
    const unsigned lds = (unsigned)(uintptr_t)(&Bsh[0]);
    const v4u g0 = {
        1u,                                            // count=1, user D#
        lds,                                           // lds_addr
        (unsigned)(ga & 0xFFFFFFFFu),                  // global_addr lo
        ((unsigned)((ga >> 32) & 0x01FFFFFFu)) | 0x80000000u  // hi | type=2
    };
    const v8i g1 = {
        (int)(1u << 16),    // data_size=2B; wg_mask=0; no pad/iterate
        (int)(256u << 16),  // tensor_dim0[15:0]=256   (bits 63:48)
        (int)(64u << 16),   // tensor_dim1[15:0]=64    (bits 95:80)
        (int)(256u << 16),  // tile_dim0=256           (bits 127:112)
        64,                 // tile_dim1=64; tile_dim2=0
        256,                // tensor_dim0_stride lo32
        0, 0
    };
    const v4i gz4 = {0, 0, 0, 0};
    const v8i gz8 = {0, 0, 0, 0, 0, 0, 0, 0};
    // 6-arg toolchain signature: (g0, g1, g2, g3, g4, cpol)
    __builtin_amdgcn_tensor_load_to_lds(g0, g1, gz4, gz4, gz8, 0);
    __builtin_amdgcn_s_wait_tensorcnt(0);
  }
#else
  for (int t = tid; t < 64 * 256; t += 256) Bsh[t] = Ub[t];
#endif

  for (int t = tid; t < 16 * 64; t += 256) {
    const int r = t >> 6, c = t & 63;
    Ash[t] = (_Float16)ws[OFF_GHS + (h0 + r) * 64 + c];
  }
  __syncthreads();

  const int n0 = wave * 32;
  v8f acc0 = {}, acc1 = {};
#pragma unroll
  for (int kb = 0; kb < 2; ++kb) {
    v16h a  = load_a16(Ash, 64, 0, kb * 32, lane);
    v16h b0 = load_b16(Bsh, 256, kb * 32, n0, lane);
    v16h b1 = load_b16(Bsh, 256, kb * 32, n0 + 16, lane);
    acc0 = wmma_f16(a, b0, acc0);
    acc1 = wmma_f16(a, b1, acc1);
  }

  float* __restrict__ O = out + (size_t)bo * HW;
#pragma unroll
  for (int r = 0; r < 8; ++r) {
    const int h = h0 + r + ((lane >> 4) << 3);
    const int w = n0 + (lane & 15);
    const float v0 = acc0[r] + O[h * 256 + w];
    O[h * 256 + w] = gelu_f(v0);
    const float v1 = acc1[r] + O[h * 256 + w + 16];
    O[h * 256 + w + 16] = gelu_f(v1);
  }
}

// ---------------------------------------------------------------------------
extern "C" void kernel_launch(void* const* d_in, const int* in_sizes, int n_in,
                              void* d_out, int out_size, void* d_ws, size_t ws_size,
                              hipStream_t stream) {
  const float* x      = (const float*)d_in[0];
  const float* w_real = (const float*)d_in[1];
  const float* w_imag = (const float*)d_in[2];
  const float* conv_w = (const float*)d_in[3];
  const float* conv_b = (const float*)d_in[4];
  float* out = (float*)d_out;
  float* ws  = (float*)d_ws;
  (void)in_sizes; (void)n_in; (void)out_size; (void)ws_size;

  k0_twiddles<<<64, 256, 0, stream>>>(ws);
  k1_rowdft  <<<2 * NBI, 256, 0, stream>>>(x, ws);
  k2_coldft  <<<NBI, 256, 0, stream>>>(ws);
  k3_mix     <<<NKL, 128, 0, stream>>>(ws, w_real, w_imag);
  k4_invcol  <<<NBO, 256, 0, stream>>>(ws);
  k5a_bypass <<<Bsz * 512, 256, 0, stream>>>(x, conv_w, conv_b, out);
  k5b_final  <<<NBO * 16, 256, 0, stream>>>(ws, out);
}
